// PointerNet_33432025432502
// MI455X (gfx1250) — matrix operand order repaired and checked by reading, not compiled, not run on hardware
//
#include <hip/hip_runtime.h>
#include <hip/hip_bf16.h>
#include <stdint.h>

// ---------------- problem dims (fixed) ----------------
#define BB  256
#define LI  512
#define LO  128
#define CC  16
#define EE  128
#define HH  256

// Flip to 0 if the async-to-LDS mnemonic is rejected by the assembler.
#define USE_ASYNC_LDS 1

typedef __attribute__((ext_vector_type(16))) __bf16 v16bf;
typedef __attribute__((ext_vector_type(8)))  float  v8f;

union AFrag { v16bf v; uint4 q[2]; __bf16 h[16]; };
union B16x8 { uint4 u; __bf16 h[8]; };

// ---------------- fast transcendentals (v_exp_f32 / v_rcp_f32) ----------------
__device__ __forceinline__ float fsig(float x) {
    return __builtin_amdgcn_rcpf(1.0f + __builtin_amdgcn_exp2f(x * -1.44269504089f));
}
__device__ __forceinline__ float ftanh(float x) {
    return 1.0f - 2.0f * __builtin_amdgcn_rcpf(1.0f + __builtin_amdgcn_exp2f(x * 2.88539008178f));
}

// ---------------- 16B copy: global -> LDS ----------------
// Async variant: GLOBAL_LOAD_ASYNC_TO_LDS_B128 (ASYNCcnt-tracked, per-lane 16B).
// VDST = per-lane LDS byte offset (low 32 bits of generic shared pointer),
// VADDR = 64-bit global address, saddr = off.
__device__ __forceinline__ void copy16_g2l(const void* g, void* lds) {
#if USE_ASYNC_LDS
    unsigned l32 = (unsigned)(size_t)lds;   // shared generic ptr: low 32 bits = LDS offset
    asm volatile("global_load_async_to_lds_b128 %0, %1, off"
                 :: "v"(l32), "v"(g) : "memory");
#else
    *(uint4*)lds = *(const uint4*)g;
#endif
}
__device__ __forceinline__ void copy_fence_async() {
#if USE_ASYNC_LDS
    asm volatile("s_wait_asynccnt 0x0" ::: "memory");
#endif
}

// ---------------- WMMA fragment loaders ----------------
// A (16x32 bf16): lane l -> row M=(l&15); elems j=0..7 -> K=kg*8+j, j=8..15 -> K=16+kg*8+(j-8).
__device__ __forceinline__ v16bf load_afrag_p(const __bf16* p) {
    AFrag f;
    f.q[0] = *(const uint4*)(p);
    f.q[1] = *(const uint4*)(p + 16);
    return f.v;
}
// B fragment from LDS staging buffer: frag (kt,q) at (kt*4+q)*512 elems, lane at +lane*16.
__device__ __forceinline__ v16bf load_bfrag_lds(const __bf16* sB, int frag, int lane) {
    const uint4* p = (const uint4*)(sB + (size_t)frag * 512 + lane * 16);
    AFrag f;
    f.q[0] = p[0];
    f.q[1] = p[1];
    return f.v;
}

#define WMMA_BF16(ACC, A, Bv) \
    (ACC) = __builtin_amdgcn_wmma_f32_16x16x32_bf16(false, (A), false, (Bv), (short)0, (ACC), false, false)

// ---------------- weight packing into B-fragment layout ----------------
// B[k][n] = Wsrc[n][k]; combined K = [0,K0) from W0, [K0,K0+K1) from W1, else 0.
// HW layout: lane l -> N=(l&15), K=(l>>4)*8 + j for j<8, 16+(l>>4)*8+(j-8) for j>=8.
__global__ void pack_B_kernel(const float* __restrict__ W0, int K0,
                              const float* __restrict__ W1, int K1,
                              int Nt, int Kt, __bf16* __restrict__ out) {
    int i = blockIdx.x * blockDim.x + threadIdx.x;
    if (i >= Kt * Nt * 32) return;
    int lane = i & 31;
    int frag = i >> 5;
    int nt = frag % Nt;
    int kt = frag / Nt;
    int n  = nt * 16 + (lane & 15);
    int kk = kt * 32 + (lane >> 4) * 8;
    __bf16* o = out + (size_t)frag * 512 + lane * 16;
#pragma unroll
    for (int j = 0; j < 16; ++j) {
        int k = kk + (j < 8 ? j : j + 8);
        float v = 0.0f;
        if (k < K0)                    v = W0[(size_t)n * K0 + k];
        else if (W1 && (k - K0) < K1)  v = W1[(size_t)n * K1 + (k - K0)];
        o[j] = (__bf16)v;
    }
}

// inputs [B,Li,C] f32 -> [B*Li, 32] bf16 (K padded 16->32)
__global__ void cvt_inputs_kernel(const float* __restrict__ in, __bf16* __restrict__ out) {
    long i = (long)blockIdx.x * blockDim.x + threadIdx.x;
    long r = i >> 5;
    int  c = (int)(i & 31);
    out[i] = (c < CC) ? (__bf16)in[r * CC + c] : (__bf16)0.0f;
}

__global__ void zero_state_kernel(__bf16* __restrict__ h, float* __restrict__ c) {
    int i = blockIdx.x * blockDim.x + threadIdx.x;
    h[i] = (__bf16)0.0f;
    c[i] = 0.0f;
}

__global__ void init_xh_kernel(__bf16* __restrict__ xh, const float* __restrict__ x0,
                               const __bf16* __restrict__ hfin) {
    int i = blockIdx.x * blockDim.x + threadIdx.x;          // 256*384
    int b = i / 384, j = i % 384;
    xh[i] = (j < EE) ? (__bf16)x0[j] : hfin[b * HH + (j - EE)];
}

// ---------------- LDS-staged generic WMMA GEMM ----------------
// Block = 8 waves sharing one B tile-column group (nt); wave w -> mt = mg*8+w.
// Grid = (M/128) * NQ blocks.
template<int KT, int NT, int OUTF32>
__global__ void gemm_staged_kernel(const __bf16* __restrict__ A, long lda,
                                   const __bf16* __restrict__ Bp,
                                   const float* __restrict__ bias,
                                   void* __restrict__ Out, long ldo) {
    constexpr int NQ = NT / 4;
    __shared__ __bf16 sB[KT * 4 * 512];
    int tid = threadIdx.x;
    int nt  = blockIdx.x % NQ;
    int mg  = blockIdx.x / NQ;

    // Stage B fragments for this nt: KT x 4 quadrant frags (1KB each).
#pragma unroll 1
    for (int idx = tid; idx < KT * 4 * 64; idx += 256) {
        int kt = idx >> 8;
        int q  = (idx >> 6) & 3;
        int c  = idx & 63;
        const uint4* src = (const uint4*)(Bp + (size_t)(kt * NT + nt + q * NQ) * 512) + c;
        ((uint4*)sB)[idx] = *src;
    }
    __syncthreads();

    int wid = tid >> 5, lane = tid & 31;
    int mt = mg * 8 + wid;
    const __bf16* aptr = A + (size_t)(mt * 16 + (lane & 15)) * lda + (lane >> 4) * 8;

    v8f acc0 = {}, acc1 = {}, acc2 = {}, acc3 = {};
#pragma unroll
    for (int kt = 0; kt < KT; ++kt) {
        v16bf a  = load_afrag_p(aptr + kt * 32);
        v16bf b0 = load_bfrag_lds(sB, kt * 4 + 0, lane);
        v16bf b1 = load_bfrag_lds(sB, kt * 4 + 1, lane);
        v16bf b2 = load_bfrag_lds(sB, kt * 4 + 2, lane);
        v16bf b3 = load_bfrag_lds(sB, kt * 4 + 3, lane);
        WMMA_BF16(acc0, a, b0);
        WMMA_BF16(acc1, a, b1);
        WMMA_BF16(acc2, a, b2);
        WMMA_BF16(acc3, a, b3);
    }

    int nl = lane & 15, mh = (lane >> 4) * 8;
#pragma unroll
    for (int q = 0; q < 4; ++q) {
        const v8f& acc = (q == 0) ? acc0 : (q == 1) ? acc1 : (q == 2) ? acc2 : acc3;
        int ncol = (nt + q * NQ) * 16 + nl;
        float bv = bias ? bias[ncol] : 0.0f;
#pragma unroll
        for (int r = 0; r < 8; ++r) {
            int m = mt * 16 + r + mh;
            float v = acc[r] + bv;
            if (OUTF32) ((float*)Out)[(size_t)m * ldo + ncol] = v;
            else        ((__bf16*)Out)[(size_t)m * ldo + ncol] = (__bf16)v;
        }
    }
}

// ---------------- fused LSTM step (LDS-staged weights, async copy) ----------------
// gates = [A0 | A1] @ packedB + bias; quadrants i,f,g,o at nt, nt+16, nt+32, nt+48.
// Grid: 32 blocks (2 mt-groups x 16 nt) x 256 threads.
template<int K0>
__global__ void lstm_step_kernel(const __bf16* __restrict__ A0, long lda0,
                                 const __bf16* __restrict__ A1, long lda1,
                                 const __bf16* __restrict__ Bp,
                                 const float* __restrict__ bias,
                                 float* __restrict__ cstate,
                                 __bf16* __restrict__ hout, long ldh,
                                 __bf16* __restrict__ hout2, long ldh2) {
    constexpr int KT = 12, NT = 64, NQ = 16;
    __shared__ __bf16 sB[KT * 4 * 512];      // 48KB of the 320KB WGP LDS
    int tid = threadIdx.x;
    int nt  = blockIdx.x % NQ;
    int mg  = blockIdx.x / NQ;

    // Async-stage the 48 fragments for this nt straight into LDS (ASYNCcnt path).
#pragma unroll 1
    for (int idx = tid; idx < KT * 4 * 64; idx += 256) {
        int kt = idx >> 8;
        int q  = (idx >> 6) & 3;
        int c  = idx & 63;
        const uint4* src = (const uint4*)(Bp + (size_t)(kt * NT + nt + q * NQ) * 512) + c;
        copy16_g2l(src, &((uint4*)sB)[idx]);
    }
    copy_fence_async();
    __syncthreads();

    int wid = tid >> 5, lane = tid & 31;
    int mt = mg * 8 + wid;
    const __bf16* a0 = A0 + (size_t)(mt * 16 + (lane & 15)) * lda0 + (lane >> 4) * 8;
    const __bf16* a1 = A1 ? A1 + (size_t)(mt * 16 + (lane & 15)) * lda1 + (lane >> 4) * 8 : nullptr;

    v8f ai = {}, af = {}, ag = {}, ao = {};
#pragma unroll
    for (int kt = 0; kt < KT; ++kt) {
        const int kc = kt * 32;
        v16bf a = (kc < K0) ? load_afrag_p(a0 + kc) : load_afrag_p(a1 + (kc - K0));
        v16bf bi = load_bfrag_lds(sB, kt * 4 + 0, lane);
        v16bf bf = load_bfrag_lds(sB, kt * 4 + 1, lane);
        v16bf bg = load_bfrag_lds(sB, kt * 4 + 2, lane);
        v16bf bo = load_bfrag_lds(sB, kt * 4 + 3, lane);
        WMMA_BF16(ai, a, bi);
        WMMA_BF16(af, a, bf);
        WMMA_BF16(ag, a, bg);
        WMMA_BF16(ao, a, bo);
    }

    int n  = nt * 16 + (lane & 15);
    int mh = (lane >> 4) * 8;
    float b_i = bias[0 * HH + n], b_f = bias[1 * HH + n];
    float b_g = bias[2 * HH + n], b_o = bias[3 * HH + n];
#pragma unroll
    for (int r = 0; r < 8; ++r) {
        int m = mt * 16 + r + mh;
        float gi = fsig(ai[r] + b_i);
        float gf = fsig(af[r] + b_f);
        float gg = ftanh(ag[r] + b_g);
        float go = fsig(ao[r] + b_o);
        float co = cstate[m * HH + n];
        float cn = gf * co + gi * gg;
        float hn = go * ftanh(cn);
        cstate[m * HH + n] = cn;
        hout[(size_t)m * ldh + n] = (__bf16)hn;
        if (hout2) hout2[(size_t)m * ldh2 + n] = (__bf16)hn;
    }
}

// ---------------- attention + argmax + gather (one block per batch row) ----------------
__global__ void attn_step_kernel(const __bf16* __restrict__ encproj,   // [B*Li, H] bf16, L2-resident
                                 const float* __restrict__ dterm,      // [B, H]
                                 const float* __restrict__ attV,       // [H]
                                 const __bf16* __restrict__ embbf,     // [B*Li, E] bf16
                                 float* __restrict__ logits,           // [B, Li, Lo]
                                 float* __restrict__ pointers,         // [B, Lo]
                                 __bf16* __restrict__ xh_next,         // [B, 384]
                                 int t) {
    __shared__ float sdt[HH];
    __shared__ float sV[HH];
    __shared__ float ssc[LI];
    __shared__ float rv[256];
    __shared__ int   ri[256];

    int b = blockIdx.x;
    int tid = threadIdx.x;
    sdt[tid] = dterm[b * HH + tid];
    sV[tid]  = attV[tid];
    __syncthreads();

    int w = tid >> 5, lane = tid & 31;
    int kb = lane * 8;
    for (int l = w; l < LI; l += 8) {
        const __bf16* row = encproj + ((size_t)(b * LI + l)) * HH + kb;
        __builtin_prefetch((const void*)(row + 8 * HH), 0, 1);   // next row for this wave
        B16x8 x;
        x.u = *(const uint4*)(row);
        float s = 0.0f;
#pragma unroll
        for (int j = 0; j < 8; ++j) {
            int k = kb + j;
            s += ftanh((float)x.h[j] + sdt[k]) * sV[k];
        }
#pragma unroll
        for (int off = 16; off > 0; off >>= 1) s += __shfl_xor(s, off, 32);
        if (lane == 0) {
            ssc[l] = s;
            logits[((size_t)(b * LI + l)) * LO + t] = s;
        }
    }
    __syncthreads();

    float v0 = ssc[tid], v1 = ssc[tid + 256];
    if (v1 > v0) { rv[tid] = v1; ri[tid] = tid + 256; }
    else         { rv[tid] = v0; ri[tid] = tid; }
    __syncthreads();
#pragma unroll
    for (int s = 128; s > 0; s >>= 1) {
        if (tid < s) {
            float a = rv[tid], c = rv[tid + s];
            if (c > a || (c == a && ri[tid + s] < ri[tid])) { rv[tid] = c; ri[tid] = ri[tid + s]; }
        }
        __syncthreads();
    }
    int best = ri[0];
    if (tid == 0) pointers[b * LO + t] = (float)best;
    if (tid < EE) xh_next[b * 384 + tid] = embbf[((size_t)(b * LI + best)) * EE + tid];
}

// ---------------- host orchestration ----------------
extern "C" void kernel_launch(void* const* d_in, const int* in_sizes, int n_in,
                              void* d_out, int out_size, void* d_ws, size_t ws_size,
                              hipStream_t stream) {
    (void)in_sizes; (void)n_in; (void)out_size; (void)ws_size;
    const float* inputs   = (const float*)d_in[0];
    const float* emb_W    = (const float*)d_in[1];
    const float* emb_b    = (const float*)d_in[2];
    const float* enc_Wih  = (const float*)d_in[3];
    const float* enc_Whh  = (const float*)d_in[4];
    const float* enc_b    = (const float*)d_in[5];
    const float* dec_Wih  = (const float*)d_in[6];
    const float* dec_Whh  = (const float*)d_in[7];
    const float* dec_b    = (const float*)d_in[8];
    const float* att_W1   = (const float*)d_in[9];
    const float* att_W2   = (const float*)d_in[10];
    const float* att_V    = (const float*)d_in[11];
    const float* dec_in0  = (const float*)d_in[12];

    uint8_t* p = (uint8_t*)d_ws;
    auto carve = [&](size_t bytes) -> void* {
        void* r = (void*)p;
        p += (bytes + 255) & ~(size_t)255;
        return r;
    };
    const size_t ML = (size_t)BB * LI;
    __bf16* pk_emb = (__bf16*)carve((size_t)1 * 8  * 512 * 2);
    __bf16* pk_enc = (__bf16*)carve((size_t)12 * 64 * 512 * 2);
    __bf16* pk_dec = (__bf16*)carve((size_t)12 * 64 * 512 * 2);
    __bf16* pk_w1  = (__bf16*)carve((size_t)8 * 16 * 512 * 2);
    __bf16* pk_w2  = (__bf16*)carve((size_t)8 * 16 * 512 * 2);
    __bf16* in_bf  = (__bf16*)carve(ML * 32 * 2);
    __bf16* emb_bf = (__bf16*)carve(ML * EE * 2);
    __bf16* ench   = (__bf16*)carve(ML * HH * 2);
    __bf16* eproj  = (__bf16*)carve(ML * HH * 2);        // bf16 -> L2-resident
    __bf16* hb0    = (__bf16*)carve((size_t)BB * HH * 2);
    __bf16* hb1    = (__bf16*)carve((size_t)BB * HH * 2);
    float*  cst    = (float*) carve((size_t)BB * HH * 4);
    __bf16* xh0    = (__bf16*)carve((size_t)BB * 384 * 2);
    __bf16* xh1    = (__bf16*)carve((size_t)BB * 384 * 2);
    float*  dterm  = (float*) carve((size_t)BB * HH * 4);

    float* logits   = (float*)d_out;
    float* pointers = (float*)d_out + (size_t)BB * LI * LO;

    // --- one-time prep ---
    pack_B_kernel<<<1, 256, 0, stream>>>(emb_W, CC, nullptr, 0, 8, 1, pk_emb);
    pack_B_kernel<<<96, 256, 0, stream>>>(enc_Wih, EE, enc_Whh, HH, 64, 12, pk_enc);
    pack_B_kernel<<<96, 256, 0, stream>>>(dec_Wih, EE, dec_Whh, HH, 64, 12, pk_dec);
    pack_B_kernel<<<16, 256, 0, stream>>>(att_W1, HH, nullptr, 0, 16, 8, pk_w1);
    pack_B_kernel<<<16, 256, 0, stream>>>(att_W2, HH, nullptr, 0, 16, 8, pk_w2);
    cvt_inputs_kernel<<<(int)(ML * 32 / 256), 256, 0, stream>>>(inputs, in_bf);
    zero_state_kernel<<<BB * HH / 256, 256, 0, stream>>>(hb0, cst);

    // --- embedding GEMM: (M/128)*NQ = 1024*2 blocks ---
    gemm_staged_kernel<1, 8, 0><<<2048, 256, 0, stream>>>(in_bf, 32, pk_emb, emb_b, emb_bf, EE);

    // --- encoder recurrence: 512 dependent steps ---
    for (int t = 0; t < LI; ++t) {
        const __bf16* hin  = (t & 1) ? hb1 : hb0;
        __bf16*       hout = (t & 1) ? hb0 : hb1;
        lstm_step_kernel<EE><<<32, 256, 0, stream>>>(
            emb_bf + (size_t)t * EE, (long)LI * EE,
            hin, HH,
            pk_enc, enc_b, cst,
            hout, HH,
            ench + (size_t)t * HH, (long)LI * HH);
    }
    // final h in hb0 (t=511 writes hb0)

    // --- enc_proj = enc_hiddens @ att_W1^T : 1024*4 blocks ---
    gemm_staged_kernel<8, 16, 0><<<4096, 256, 0, stream>>>(ench, HH, pk_w1, nullptr, eproj, HH);

    init_xh_kernel<<<BB * 384 / 256, 256, 0, stream>>>(xh0, dec_in0, hb0);

    // --- decoder: 128 dependent steps ---
    for (int t = 0; t < LO; ++t) {
        __bf16* xin   = (t & 1) ? xh1 : xh0;
        __bf16* xnext = (t & 1) ? xh0 : xh1;
        lstm_step_kernel<384><<<32, 256, 0, stream>>>(
            xin, 384,
            nullptr, 0,
            pk_dec, dec_b, cst,
            xnext + EE, 384, nullptr, 0);
        // dec_term = h @ att_W2^T : (16/8)*4 = 8 blocks, f32 out
        gemm_staged_kernel<8, 16, 1><<<8, 256, 0, stream>>>(xnext + EE, 384, pk_w2, nullptr, dterm, HH);
        attn_step_kernel<<<BB, 256, 0, stream>>>(eproj, dterm, att_V, emb_bf,
                                                 logits, pointers, xnext, t);
    }
}
